// MultiHeadAttention_80479097193004
// MI455X (gfx1250) — compile-verified, matrix operand order checked by reading, not run on hardware
//
#include <hip/hip_runtime.h>
#include <hip/hip_bf16.h>

// ---------------------------------------------------------------------------
// MHA forward for MI455X (gfx1250, wave32, WMMA).
// Pipeline: fp32->bf16 cast -> 3 projection GEMMs (bf16 WMMA, f32 acc)
//           -> flash attention (bf16 WMMA QK^T and PV, online softmax)
//           -> output projection GEMM (f32 result).
// GEMM wave tile: 32(M) x 64(N) -> 8 v_wmma per 32-deep k-step, 6 frag loads.
// ---------------------------------------------------------------------------

typedef __attribute__((ext_vector_type(16))) __bf16 v16bf;
typedef __attribute__((ext_vector_type(8)))  float  v8f;

constexpr int B_ = 2, S_ = 2048, E_ = 1024, H_ = 16, D_ = 64;
constexpr int M_ = B_ * S_;   // 4096 token rows

struct Frag2 { uint4 a, b; }; // 32 bytes == v16bf

static __device__ __forceinline__ unsigned short f2bf(float x) {
    unsigned u = __builtin_bit_cast(unsigned, x);
    unsigned r = u + 0x7FFFu + ((u >> 16) & 1u);   // round-to-nearest-even
    return (unsigned short)(r >> 16);
}

// Load a 16x32 bf16 A/B WMMA fragment from a row-major bf16 (ushort) array.
// Per ISA 16-bit layout: lane&15 selects row/col; lanes>=16 take K offsets +8;
// each lane holds K runs [k+off, k+off+8) and [k+off+16, k+off+24).
template <typename PT>
static __device__ __forceinline__ v16bf load_frag(const PT* base, size_t row,
                                                  int ldk, int k, int hi) {
    const PT* p = base + row * (size_t)ldk + (size_t)k + (size_t)(hi << 3);
    Frag2 fr;
    fr.a = *reinterpret_cast<const uint4*>(p);
    fr.b = *reinterpret_cast<const uint4*>(p + 16);
    return __builtin_bit_cast(v16bf, fr);
}

static __device__ __forceinline__ v8f wmma_bf16(v16bf a, v16bf b, v8f c) {
    return __builtin_amdgcn_wmma_f32_16x16x32_bf16(false, a, false, b,
                                                   (short)0, c, false, false);
}

// ---------------------------------------------------------------------------
// fp32 -> bf16 cast, two elements per thread (packed 32-bit stores)
// ---------------------------------------------------------------------------
__global__ __launch_bounds__(256) void cvt_bf16_kernel(
    const float* __restrict__ src, unsigned int* __restrict__ dst, int npair) {
    int i = blockIdx.x * 256 + threadIdx.x;
    if (i < npair) {
        float2 f = reinterpret_cast<const float2*>(src)[i];
        dst[i] = (unsigned)f2bf(f.x) | ((unsigned)f2bf(f.y) << 16);
    }
}

// ---------------------------------------------------------------------------
// GEMM: C[M,N] = A[M,K] * W[N,K]^T   (torch Linear convention)
// A, W are bf16 (as ushort). 4 waves/block, each wave owns a 32x64 tile.
// mode 0: f32 row-major out      (final projection -> d_out)
// mode 1: bf16 out [B,H,S,D]     (Q, K)
// mode 2: bf16 out [B,H,D,S]     (V transposed for PV fragment loads)
// ---------------------------------------------------------------------------
__global__ __launch_bounds__(128) void wmma_gemm_kernel(
    const unsigned short* __restrict__ A, const unsigned short* __restrict__ W,
    void* __restrict__ out, int M, int N, int K, int mode) {
    const int lane   = threadIdx.x & 31;
    const int wave   = threadIdx.x >> 5;
    const int lane15 = lane & 15;
    const int hi     = lane >> 4;
    const int m0 = blockIdx.y * 128 + wave * 32;
    const int n0 = blockIdx.x * 64;

    v8f acc[2][4] = {{v8f{}, v8f{}, v8f{}, v8f{}},
                     {v8f{}, v8f{}, v8f{}, v8f{}}};

    for (int k = 0; k < K; k += 32) {
        v16bf a0 = load_frag(A, (size_t)(m0 +  0 + lane15), K, k, hi);
        v16bf a1 = load_frag(A, (size_t)(m0 + 16 + lane15), K, k, hi);
        v16bf b0 = load_frag(W, (size_t)(n0 +  0 + lane15), K, k, hi);
        v16bf b1 = load_frag(W, (size_t)(n0 + 16 + lane15), K, k, hi);
        v16bf b2 = load_frag(W, (size_t)(n0 + 32 + lane15), K, k, hi);
        v16bf b3 = load_frag(W, (size_t)(n0 + 48 + lane15), K, k, hi);
        if (k + 256 < K) {  // pull a future A tile toward the caches
            __builtin_prefetch(A + (size_t)(m0 + lane15) * K + k + 256, 0, 1);
        }
        acc[0][0] = wmma_bf16(a0, b0, acc[0][0]);
        acc[0][1] = wmma_bf16(a0, b1, acc[0][1]);
        acc[0][2] = wmma_bf16(a0, b2, acc[0][2]);
        acc[0][3] = wmma_bf16(a0, b3, acc[0][3]);
        acc[1][0] = wmma_bf16(a1, b0, acc[1][0]);
        acc[1][1] = wmma_bf16(a1, b1, acc[1][1]);
        acc[1][2] = wmma_bf16(a1, b2, acc[1][2]);
        acc[1][3] = wmma_bf16(a1, b3, acc[1][3]);
    }

    // C layout: vgpr i, lane -> (m = 16*r + i + 8*hi, n = 16*t + lane15)
#pragma unroll
    for (int r = 0; r < 2; ++r) {
#pragma unroll
        for (int t = 0; t < 4; ++t) {
#pragma unroll
            for (int i = 0; i < 8; ++i) {
                const int m = m0 + r * 16 + i + 8 * hi;
                const int n = n0 + t * 16 + lane15;
                const float v = acc[r][t][i];
                if (mode == 0) {
                    reinterpret_cast<float*>(out)[(size_t)m * N + n] = v;
                } else {
                    const int b = m >> 11, s = m & (S_ - 1);
                    const int h = n >> 6,  d = n & (D_ - 1);
                    const size_t idx = (mode == 1)
                        ? ((((size_t)b * H_ + h) * S_ + s) * D_ + d)
                        : ((((size_t)b * H_ + h) * D_ + d) * S_ + s);
                    reinterpret_cast<unsigned short*>(out)[idx] = f2bf(v);
                }
            }
        }
    }
}

// ---------------------------------------------------------------------------
// Flash attention, causal. One wave (block of 32) per 16-query tile per head.
// Q,K: [B,H,S,D] bf16; Vt: [B,H,D,S] bf16; ctx out: [B,S,E] bf16.
// ---------------------------------------------------------------------------
__global__ __launch_bounds__(32) void attn_kernel(
    const unsigned short* __restrict__ Q, const unsigned short* __restrict__ Km,
    const unsigned short* __restrict__ Vt, unsigned short* __restrict__ ctx) {
    __shared__ __align__(16) unsigned short Plds[16 * 32];

    const int lane   = threadIdx.x & 31;
    const int lane15 = lane & 15;
    const int hi     = lane >> 4;
    const int q0 = blockIdx.x * 16;
    const int h  = blockIdx.y;
    const int b  = blockIdx.z;

    const unsigned short* Qh = Q  + (((size_t)b * H_ + h) * S_) * D_;
    const unsigned short* Kh = Km + (((size_t)b * H_ + h) * S_) * D_;
    const unsigned short* Vh = Vt + (((size_t)b * H_ + h) * D_) * S_;

    const v16bf aq0 = load_frag(Qh, (size_t)(q0 + lane15), D_, 0,  hi);
    const v16bf aq1 = load_frag(Qh, (size_t)(q0 + lane15), D_, 32, hi);

    v8f o[4] = {v8f{}, v8f{}, v8f{}, v8f{}};
    float mrow[8], lrow[8];
#pragma unroll
    for (int i = 0; i < 8; ++i) { mrow[i] = -__builtin_inff(); lrow[i] = 0.f; }

    const int qlast = q0 + 15;
    const float scl = 0.125f;   // 1/sqrt(64)

    for (int k0 = 0; k0 <= qlast; k0 += 32) {
        // S tiles: rows=16 queries, cols=32 keys (two 16-col subtiles)
        v8f s0 = {}, s1 = {};
        {
            v16bf bk = load_frag(Kh, (size_t)(k0 + lane15), D_, 0, hi);
            s0 = wmma_bf16(aq0, bk, s0);
            bk = load_frag(Kh, (size_t)(k0 + lane15), D_, 32, hi);
            s0 = wmma_bf16(aq1, bk, s0);
        }
        {
            v16bf bk = load_frag(Kh, (size_t)(k0 + 16 + lane15), D_, 0, hi);
            s1 = wmma_bf16(aq0, bk, s1);
            bk = load_frag(Kh, (size_t)(k0 + 16 + lane15), D_, 32, hi);
            s1 = wmma_bf16(aq1, bk, s1);
        }

        // online softmax update; row r = i + 8*hi lives in 16-lane half-wave
#pragma unroll
        for (int i = 0; i < 8; ++i) {
            const int qi = q0 + i + 8 * hi;
            float v0 = (k0 + lane15      <= qi) ? s0[i] * scl : -__builtin_inff();
            float v1 = (k0 + 16 + lane15 <= qi) ? s1[i] * scl : -__builtin_inff();
            float rm = fmaxf(v0, v1);
            rm = fmaxf(rm, __shfl_xor(rm, 8));
            rm = fmaxf(rm, __shfl_xor(rm, 4));
            rm = fmaxf(rm, __shfl_xor(rm, 2));
            rm = fmaxf(rm, __shfl_xor(rm, 1));
            const float mn = fmaxf(mrow[i], rm);
            const float sc = __expf(mrow[i] - mn);
            const float p0 = __expf(v0 - mn);
            const float p1 = __expf(v1 - mn);
            float rs = p0 + p1;
            rs += __shfl_xor(rs, 8);
            rs += __shfl_xor(rs, 4);
            rs += __shfl_xor(rs, 2);
            rs += __shfl_xor(rs, 1);
            lrow[i] = lrow[i] * sc + rs;
            mrow[i] = mn;
            o[0][i] *= sc; o[1][i] *= sc; o[2][i] *= sc; o[3][i] *= sc;
            const int r = i + 8 * hi;
            Plds[r * 32 + lane15]      = f2bf(p0);
            Plds[r * 32 + 16 + lane15] = f2bf(p1);
        }

        __builtin_amdgcn_wave_barrier();
        asm volatile("s_wait_dscnt 0" ::: "memory");

        // re-read P in A-fragment layout (16x32, ld=32) from LDS
        v16bf pf;
        {
            Frag2 fr;
            fr.a = *reinterpret_cast<const uint4*>(&Plds[lane15 * 32 + (hi << 3)]);
            fr.b = *reinterpret_cast<const uint4*>(&Plds[lane15 * 32 + 16 + (hi << 3)]);
            pf = __builtin_bit_cast(v16bf, fr);
        }

        // O[16,64] += P[16,32] * V[32,64]; B-frag col n reads Vt row n (contig S)
#pragma unroll
        for (int t = 0; t < 4; ++t) {
            v16bf vf = load_frag(Vh, (size_t)(t * 16 + lane15), S_, k0, hi);
            o[t] = wmma_bf16(pf, vf, o[t]);
        }
        __builtin_amdgcn_wave_barrier();
        asm volatile("" ::: "memory");
    }

    // normalize and emit ctx[B,S,E] bf16 (E index = h*64 + d)
#pragma unroll
    for (int i = 0; i < 8; ++i) {
        const float inv = 1.0f / lrow[i];
        const int s = q0 + i + 8 * hi;
#pragma unroll
        for (int t = 0; t < 4; ++t) {
            const int e = h * D_ + t * 16 + lane15;
            ctx[((size_t)b * S_ + s) * E_ + e] = f2bf(o[t][i] * inv);
        }
    }
}

// ---------------------------------------------------------------------------
extern "C" void kernel_launch(void* const* d_in, const int* in_sizes, int n_in,
                              void* d_out, int out_size, void* d_ws, size_t ws_size,
                              hipStream_t stream) {
    (void)in_sizes; (void)n_in; (void)out_size; (void)ws_size;
    const float* q  = (const float*)d_in[0];
    const float* k  = (const float*)d_in[1];
    const float* v  = (const float*)d_in[2];
    const float* wq = (const float*)d_in[3];
    const float* wk = (const float*)d_in[4];
    const float* wv = (const float*)d_in[5];
    const float* wo = (const float*)d_in[6];
    float* out = (float*)d_out;

    // bf16 workspace carve-up (64 MiB total)
    unsigned short* p   = (unsigned short*)d_ws;
    unsigned short* xq  = p; p += (size_t)M_ * E_;
    unsigned short* xk  = p; p += (size_t)M_ * E_;
    unsigned short* xv  = p; p += (size_t)M_ * E_;
    unsigned short* bwq = p; p += (size_t)E_ * E_;
    unsigned short* bwk = p; p += (size_t)E_ * E_;
    unsigned short* bwv = p; p += (size_t)E_ * E_;
    unsigned short* bwo = p; p += (size_t)E_ * E_;
    unsigned short* Qp  = p; p += (size_t)M_ * E_;
    unsigned short* Kp  = p; p += (size_t)M_ * E_;
    unsigned short* Vtp = p; p += (size_t)M_ * E_;
    unsigned short* ctx = p; p += (size_t)M_ * E_;

    auto cvt = [&](const float* s, unsigned short* d, size_t n) {
        int npair = (int)(n / 2);
        cvt_bf16_kernel<<<(npair + 255) / 256, 256, 0, stream>>>(
            s, (unsigned int*)d, npair);
    };
    cvt(q,  xq,  (size_t)M_ * E_);
    cvt(k,  xk,  (size_t)M_ * E_);
    cvt(v,  xv,  (size_t)M_ * E_);
    cvt(wq, bwq, (size_t)E_ * E_);
    cvt(wk, bwk, (size_t)E_ * E_);
    cvt(wv, bwv, (size_t)E_ * E_);
    cvt(wo, bwo, (size_t)E_ * E_);

    dim3 ggrid(E_ / 64, M_ / 128);  // (16, 32), 128 threads = 4 waves/block
    wmma_gemm_kernel<<<ggrid, 128, 0, stream>>>(xq, bwq, Qp,  M_, E_, E_, 1);
    wmma_gemm_kernel<<<ggrid, 128, 0, stream>>>(xk, bwk, Kp,  M_, E_, E_, 1);
    wmma_gemm_kernel<<<ggrid, 128, 0, stream>>>(xv, bwv, Vtp, M_, E_, E_, 2);

    attn_kernel<<<dim3(S_ / 16, H_, B_), 32, 0, stream>>>(Qp, Kp, Vtp, ctx);

    wmma_gemm_kernel<<<ggrid, 128, 0, stream>>>(ctx, bwo, out, M_, E_, E_, 0);
}
